// TestLayer_31001073942736
// MI455X (gfx1250) — matrix operand order repaired and checked by reading, not compiled, not run on hardware
//
#include <hip/hip_runtime.h>

#define NEG_SLOPE 0.2f

typedef __attribute__((ext_vector_type(16))) __bf16 v16bf;
typedef __attribute__((ext_vector_type(8)))  float  v8f;

// ---------------------------------------------------------------------------
// Pack a weight matrix into a per-lane WMMA B-fragment layout (bf16, contiguous k, same order as the A loads).
// Output index: ((kb*NTILES + nt)*32 + lane)*16 + e, where e = 2*v + j and
//   k = kb*32 + 16*(lane>>4) + 2*v + j,  n = nt*16 + (lane&15)
// rowMajorKxN==1: W is KxN (x@W weights); ==0: W is NxK (fcW, out = acc@fcW^T)
// ---------------------------------------------------------------------------
__global__ void pack_B_bf16(const float* __restrict__ W, __bf16* __restrict__ Bp,
                            int K, int N, int rowMajorKxN)
{
    const int idx = blockIdx.x * blockDim.x + threadIdx.x;
    if (idx >= K * N) return;
    const int e  = idx & 15;
    const int L  = (idx >> 4) & 31;
    const int t  = idx >> 9;                 // kb*ntiles + nt
    const int ntiles = N >> 4;
    const int kb = t / ntiles, nt = t % ntiles;
    const int half = L >> 4, r = L & 15;
    const int k = kb * 32 + 16 * half + 2 * (e >> 1) + (e & 1);
    const int n = nt * 16 + r;
    const float val = rowMajorKxN ? W[(size_t)k * N + n] : W[(size_t)n * K + k];
    Bp[idx] = (__bf16)val;
}

// ---------------------------------------------------------------------------
// One wave computes a 16x128 output strip: 8 accumulators, 8 WMMAs per K-step.
// A (MxK f32, row-major) fragment loaded once per K-step, reused 8x.
// B comes pre-packed in bf16 fragment layout: one 32B vector load per tile.
// ---------------------------------------------------------------------------
template <int K, bool ACT>
__global__ void wmma_gemm_bf16(const float* __restrict__ A,
                               const __bf16* __restrict__ Bp,
                               const float* __restrict__ bias,
                               float* __restrict__ C, int M)
{
    constexpr int N = 128, NTILES = 8, KB = K / 32;
    const int wave = (int)((blockIdx.x * (unsigned)blockDim.x + threadIdx.x) >> 5);
    const int lane = threadIdx.x & 31;
    if (wave >= (M >> 4)) return;            // wave-uniform: EXEC stays all-1s
    const int half = lane >> 4, r = lane & 15;
    const int m0 = wave << 4;

    v8f c[NTILES];
    #pragma unroll
    for (int nt = 0; nt < NTILES; ++nt) c[nt] = (v8f){};

    #pragma unroll
    for (int kb = 0; kb < KB; ++kb) {
        // A fragment: one contiguous 16-float chunk per lane (k = 16*half + e) -> 4x global_load_b128
        const float* arow = A + (size_t)(m0 + r) * K + kb * 32;
        v16bf a;
        #pragma unroll
        for (int v = 0; v < 8; ++v) {
            const int ka = 16 * half + 2 * v;
            a[2 * v]     = (__bf16)arow[ka];
            a[2 * v + 1] = (__bf16)arow[ka + 1];
        }
        const __bf16* bp = Bp + ((size_t)kb * NTILES * 32 + lane) * 16;
        #pragma unroll
        for (int nt = 0; nt < NTILES; ++nt) {
            const v16bf b = *(const v16bf*)(bp + (size_t)nt * 512);
            c[nt] = __builtin_amdgcn_wmma_f32_16x16x32_bf16(
                        false, a, false, b, (short)0, c[nt], false, false);
        }
    }

    // C/D layout: VGPR v holds row 8*half+v, column = nt*16 + (lane&15)
    #pragma unroll
    for (int nt = 0; nt < NTILES; ++nt) {
        const float bv = ACT ? bias[nt * 16 + r] : 0.0f;
        float* crow = C + (size_t)m0 * N + nt * 16 + r;
        #pragma unroll
        for (int v = 0; v < 8; ++v) {
            float val = c[nt][v] + bv;
            if (ACT) val = (val > 0.0f) ? val : NEG_SLOPE * val;
            crow[(size_t)(8 * half + v) * N] = val;
        }
    }
}

// ---------------------------------------------------------------------------
// One wave per edge: gather 128-float message row (float4/lane), atomic-add.
// acc0+acc3 = 76.8 MB -> resident in 192 MB L2; atomics are L2-local RMWs.
// ---------------------------------------------------------------------------
__global__ void scatter_add_128(const float* __restrict__ msg,
                                const int* __restrict__ dst,
                                const int* __restrict__ src,
                                float* __restrict__ acc, int nnz)
{
    const long long tid = (long long)blockIdx.x * blockDim.x + threadIdx.x;
    const int edge = (int)(tid >> 5);
    const int lane = (int)(tid & 31);
    if (edge >= nnz) return;
    const int s = src[edge];
    const int d = dst[edge];
    const float4 v = ((const float4*)(msg + (size_t)s * 128))[lane];
    float* p = acc + (size_t)d * 128 + (size_t)lane * 4;
    atomicAdd(p + 0, v.x);
    atomicAdd(p + 1, v.y);
    atomicAdd(p + 2, v.z);
    atomicAdd(p + 3, v.w);
}

template <int K, bool ACT>
static inline void launch_gemm(const float* A, const __bf16* Bp, const float* bias,
                               float* C, int M, hipStream_t stream)
{
    const int blocks = (M / 16 * 32 + 255) / 256;
    wmma_gemm_bf16<K, ACT><<<blocks, 256, 0, stream>>>(A, Bp, bias, C, M);
}

static inline void launch_pack(const float* W, __bf16* Bp, int K, int N, int kxn,
                               hipStream_t stream)
{
    pack_B_bf16<<<(K * N + 255) / 256, 256, 0, stream>>>(W, Bp, K, N, kxn);
}

static inline void launch_scatter(const float* msg, const int* dst, const int* src,
                                  float* acc, int nnz, hipStream_t stream)
{
    const long long th = (long long)nnz * 32;
    scatter_add_128<<<(int)((th + 255) / 256), 256, 0, stream>>>(msg, dst, src, acc, nnz);
}

extern "C" void kernel_launch(void* const* d_in, const int* in_sizes, int n_in,
                              void* d_out, int out_size, void* d_ws, size_t ws_size,
                              hipStream_t stream)
{
    const int DIN = 64, DMID = 128;
    const float* x0 = (const float*)d_in[0];
    const float* x1 = (const float*)d_in[1];
    const float* x2 = (const float*)d_in[2];
    const float* x3 = (const float*)d_in[3];
    const float* x4 = (const float*)d_in[4];
    const int* adj0   = (const int*)d_in[5];   // [2][E0]: row0 = dst, row1 = src
    const int* adj3   = (const int*)d_in[6];
    const int* inc01t = (const int*)d_in[7];
    const int* inc01s = (const int*)d_in[8];
    const int* inc23t = (const int*)d_in[9];
    const int* inc23s = (const int*)d_in[10];
    const float* W_hbs0 = (const float*)d_in[11];
    const float* W_hbs3 = (const float*)d_in[12];
    const float* W_s01  = (const float*)d_in[13];
    const float* W_t23  = (const float*)d_in[14];
    const float* fc0_W  = (const float*)d_in[15];
    const float* fc0_b  = (const float*)d_in[16];
    const float* fc3_W  = (const float*)d_in[17];
    const float* fc3_b  = (const float*)d_in[18];

    const int N0 = in_sizes[0] / DIN;          // 100000
    const int N1 = in_sizes[1] / DIN;          // 300000
    const int N2 = in_sizes[2] / DIN;          // 150000
    const int N3 = in_sizes[3] / DIN;          // 50000
    const int N4 = in_sizes[4] / DIN;          // 10000
    const int E0  = in_sizes[5] / 2;           // 800000
    const int E3  = in_sizes[6] / 2;           // 800000
    const int E01 = in_sizes[7];               // 600000
    const int E23 = in_sizes[9];               // 200000

    // Workspace: msg0|msg1|msg2|msg3|acc0|acc3 (f32, ~384 MB) then packed bf16 weights
    float* ws   = (float*)d_ws;
    float* msg0 = ws;
    float* msg1 = msg0 + (size_t)N0 * DMID;
    float* msg2 = msg1 + (size_t)N1 * DMID;
    float* msg3 = msg2 + (size_t)N2 * DMID;
    float* acc0 = msg3 + (size_t)N3 * DMID;
    float* acc3 = acc0 + (size_t)N0 * DMID;
    __bf16* Bp_hbs0 = (__bf16*)(acc3 + (size_t)N3 * DMID);
    __bf16* Bp_s01  = Bp_hbs0 + (size_t)DIN * DMID;
    __bf16* Bp_t23  = Bp_s01  + (size_t)DIN * DMID;
    __bf16* Bp_hbs3 = Bp_t23  + (size_t)DIN * DMID;
    __bf16* Bp_fc0  = Bp_hbs3 + (size_t)DIN * DMID;
    __bf16* Bp_fc3  = Bp_fc0  + (size_t)DMID * DMID;

    // Stage 0: pack weights into WMMA B-fragment layout (tiny, L2-hot afterwards)
    launch_pack(W_hbs0, Bp_hbs0, DIN,  DMID, 1, stream);
    launch_pack(W_s01,  Bp_s01,  DIN,  DMID, 1, stream);
    launch_pack(W_t23,  Bp_t23,  DIN,  DMID, 1, stream);
    launch_pack(W_hbs3, Bp_hbs3, DIN,  DMID, 1, stream);
    launch_pack(fc0_W,  Bp_fc0,  DMID, DMID, 0, stream);   // fcW is NxK (out@fcW^T)
    launch_pack(fc3_W,  Bp_fc3,  DMID, DMID, 0, stream);

    // Stage 1: per-node messages, msg_r = x_r @ W (bf16 WMMA, f32 accum)
    launch_gemm<64, false>(x0, Bp_hbs0, nullptr, msg0, N0, stream);
    launch_gemm<64, false>(x1, Bp_s01,  nullptr, msg1, N1, stream);
    launch_gemm<64, false>(x2, Bp_t23,  nullptr, msg2, N2, stream);
    launch_gemm<64, false>(x3, Bp_hbs3, nullptr, msg3, N3, stream);

    // Stage 2: zero accumulators (replay-safe), then edge scatter-adds
    hipMemsetAsync(acc0, 0, (size_t)N0 * DMID * sizeof(float), stream);
    hipMemsetAsync(acc3, 0, (size_t)N3 * DMID * sizeof(float), stream);
    launch_scatter(msg0, adj0,   adj0 + E0, acc0, E0,  stream); // x_0_to_0
    launch_scatter(msg1, inc01t, inc01s,    acc0, E01, stream); // x_1_to_0
    launch_scatter(msg3, adj3,   adj3 + E3, acc3, E3,  stream); // x_3_to_3
    launch_scatter(msg2, inc23s, inc23t,    acc3, E23, stream); // x_2_to_3 (dst=inc23_s)

    // Output tuple layout: x_0_new | x_1 | x_2 | x_3_new | x_4
    float* out    = (float*)d_out;
    float* out_x0 = out;
    float* out_x1 = out_x0 + (size_t)N0 * DMID;
    float* out_x2 = out_x1 + (size_t)N1 * DIN;
    float* out_x3 = out_x2 + (size_t)N2 * DIN;
    float* out_x4 = out_x3 + (size_t)N3 * DMID;

    // Stage 3: fused FC + bias + LeakyReLU
    launch_gemm<128, true>(acc0, Bp_fc0, fc0_b, out_x0, N0, stream);
    launch_gemm<128, true>(acc3, Bp_fc3, fc3_b, out_x3, N3, stream);

    // Passthrough outputs
    hipMemcpyAsync(out_x1, x1, (size_t)N1 * DIN * sizeof(float), hipMemcpyDeviceToDevice, stream);
    hipMemcpyAsync(out_x2, x2, (size_t)N2 * DIN * sizeof(float), hipMemcpyDeviceToDevice, stream);
    hipMemcpyAsync(out_x4, x4, (size_t)N4 * DIN * sizeof(float), hipMemcpyDeviceToDevice, stream);
}
